// GatedChronologicalScenePooler_72670846648626
// MI455X (gfx1250) — matrix operand order repaired and checked by reading, not compiled, not run on hardware
//
#include <hip/hip_runtime.h>
#include <hip/hip_bf16.h>
#include <math.h>

typedef __attribute__((ext_vector_type(16))) _Float16 v16h;
typedef __attribute__((ext_vector_type(8)))  _Float16 h8;
typedef __attribute__((ext_vector_type(4)))  _Float16 h4;
typedef __attribute__((ext_vector_type(8)))  float    v8f;

#define BSZ  32
#define SSZ  2048
#define DSZ  1024
#define NHSZ 512            // D/2
#define ROWS (BSZ * SSZ)    // 65536
#define MBR  64             // rows per block in gate kernel (4 waves x 16)
#define NT   16
#define KT   32

// ---------------------------------------------------------------------------
// q' = key_w @ attention_query   (removes the 137 GFLOP keys GEMM; the q.key_b
// term is constant over s and cancels in the softmax)
// ---------------------------------------------------------------------------
__global__ void qprime_kernel(const float* __restrict__ key_w,
                              const float* __restrict__ query,
                              float* __restrict__ qprime) {
  int d = blockIdx.x * blockDim.x + threadIdx.x;
  if (d >= DSZ) return;
  const float4* kr = (const float4*)(key_w + (size_t)d * DSZ);
  const float4* qr = (const float4*)query;
  float s = 0.f;
#pragma unroll 4
  for (int i = 0; i < DSZ / 4; ++i) {
    float4 a = kr[i], b = qr[i];
    s += a.x * b.x + a.y * b.y + a.z * b.z + a.w * b.w;
  }
  qprime[d] = s;
}

// ---------------------------------------------------------------------------
// W1 [D, NH] fp32 row-major  ->  w1t [NH, D] f16 (column-major of W1)
// ---------------------------------------------------------------------------
__global__ void w1t_kernel(const float* __restrict__ w1,
                           _Float16* __restrict__ w1t) {
  int idx = blockIdx.x * blockDim.x + threadIdx.x;  // over NH*D
  if (idx >= NHSZ * DSZ) return;
  int k = idx & (DSZ - 1);
  int n = idx >> 10;
  w1t[idx] = (_Float16)w1[(size_t)k * NHSZ + n];
}

__device__ __forceinline__ void wait_async0() {
#if __has_builtin(__builtin_amdgcn_s_wait_asynccnt)
  __builtin_amdgcn_s_wait_asynccnt(0);
#else
  asm volatile("s_wait_asynccnt 0" ::: "memory");
#endif
}

// ---------------------------------------------------------------------------
// Fused gate GEMM (WMMA f16, fp32 accum) + exact GELU + w2 dot + sigmoid
// + logit = gate * (x . q') / sqrt(D)
// ---------------------------------------------------------------------------
__global__ __launch_bounds__(128) void gate_kernel(
    const float* __restrict__ x,       // [ROWS, D]
    const _Float16* __restrict__ w1t,  // [NH, D]
    const float* __restrict__ b1,      // [NH]
    const float* __restrict__ w2,      // [NH]
    const float* __restrict__ b2,      // [1]
    const float* __restrict__ qprime,  // [D]
    float* __restrict__ gate_out,      // [ROWS]
    float* __restrict__ logit_out)     // [ROWS]
{
  __shared__ __align__(16) char smem[MBR * DSZ * 2 + NT * DSZ * 2];  // 128K+32K
  _Float16* sX = (_Float16*)smem;
  _Float16* sW = (_Float16*)(smem + MBR * DSZ * 2);
  float*    sQ = (float*)(smem + MBR * DSZ * 2);  // aliases sW, used first

  const int tid  = threadIdx.x;
  const int lane = tid & 31;
  const int wave = tid >> 5;
  const int lm   = lane & 15;
  const int hi   = lane >> 4;  // 0 or 1 (lane half)
  const size_t rowBase = (size_t)blockIdx.x * MBR;

  // LDS byte offset of sW (generic LDS pointer: addr[31:0] is the LDS offset)
  const unsigned sW_off = (unsigned)(size_t)(void*)sW;
  const unsigned long long w1u64 = (unsigned long long)(size_t)w1t;

  // ---- stage x block fp32 -> f16 in LDS ----
  const float4* xg = (const float4*)(x + rowBase * DSZ);
  for (int i = tid; i < MBR * DSZ / 4; i += 128) {
    float4 v = xg[i];
    h4 hv = {(_Float16)v.x, (_Float16)v.y, (_Float16)v.z, (_Float16)v.w};
    *(h4*)(sX + 4 * i) = hv;
  }
  for (int i = tid; i < DSZ; i += 128) sQ[i] = qprime[i];
  __syncthreads();

  // ---- per-row dot(x, q') ; keep the 8 rows this lane-half will write ----
  float rd[8];
#pragma unroll
  for (int r = 0; r < 16; ++r) {
    const _Float16* xr = sX + (wave * 16 + r) * DSZ + lane * 32;
    const float* qv = sQ + lane * 32;
    float p = 0.f;
#pragma unroll
    for (int j = 0; j < 32; ++j) p += (float)xr[j] * qv[j];
#pragma unroll
    for (int m = 16; m >= 1; m >>= 1) p += __shfl_xor(p, m, 32);
    if ((r >> 3) == hi) rd[r & 7] = p;   // constant index after unroll
  }
  __syncthreads();  // done with sQ; sW region may now be staged

  float gl[8] = {0, 0, 0, 0, 0, 0, 0, 0};

  for (int nt = 0; nt < NHSZ / NT; ++nt) {
    // ---- stage 16 columns of W1 (all K, 32 KB) via async global->LDS DMA:
    // GVS mode: mem = SADDR(w1t) + VADDR(32b offset); LDS dst = VDST vgpr.
    {
      const unsigned base = (unsigned)nt * (NT * DSZ * 2u);  // byte offset in w1t
      unsigned voff = base + (unsigned)tid * 16u;
      unsigned ldsa = sW_off + (unsigned)tid * 16u;
#pragma unroll
      for (int i = 0; i < 16; ++i) {
        asm volatile("global_load_async_to_lds_b128 %0, %1, %2"
                     :
                     : "v"(ldsa + (unsigned)i * 2048u),
                       "v"(voff + (unsigned)i * 2048u),
                       "s"(w1u64)
                     : "memory");
      }
      wait_async0();
    }
    __syncthreads();

    v8f acc = {};
#pragma unroll 4
    for (int kt = 0; kt < DSZ / KT; ++kt) {
      const int k0 = kt * KT;
      // A-frag (16x32 f16): lanes 0-15 rows M=lane, K = k0+{0..7,16..23};
      // lanes 16-31 rows M=lane-16, K = k0+{8..15,24..31}
      const _Float16* ap = sX + (wave * 16 + lm) * DSZ + k0 + hi * 8;
      h8 a0 = *(const h8*)ap;
      h8 a1 = *(const h8*)(ap + 16);
      v16h A = __builtin_shufflevector(a0, a1, 0, 1, 2, 3, 4, 5, 6, 7,
                                       8, 9, 10, 11, 12, 13, 14, 15);
      // B-frag (32x16 f16): lane column N=lm, K = k0 + hi*8 + {0..7,16..23}
      const _Float16* bp = sW + lm * DSZ + k0 + hi * 8;
      h8 b0v = *(const h8*)bp;
      h8 b1v = *(const h8*)(bp + 16);
      v16h Bf = __builtin_shufflevector(b0v, b1v, 0, 1, 2, 3, 4, 5, 6, 7,
                                        8, 9, 10, 11, 12, 13, 14, 15);
      acc = __builtin_amdgcn_wmma_f32_16x16x32_f16(
          false, A, false, Bf, (short)0, acc, false, false);
    }
    // epilogue: exact GELU then w2 accumulation.
    // D-layout: VGPR r -> row (r + hi*8), col N = lm
    const int n = nt * NT + lm;
    const float bn = b1[n];
    const float wn = w2[n];
#pragma unroll
    for (int r = 0; r < 8; ++r) {
      float hv = acc[r] + bn;
      float g = 0.5f * hv * (1.f + erff(hv * 0.70710678118654752f));
      gl[r] += g * wn;
    }
    __syncthreads();  // all waves done reading sW before restaging
  }

  // reduce gate-logit across the 16 lanes sharing a row
#pragma unroll
  for (int r = 0; r < 8; ++r) {
    float v = gl[r];
#pragma unroll
    for (int m = 8; m >= 1; m >>= 1) v += __shfl_xor(v, m, 32);
    gl[r] = v;
  }
  const float bias2 = b2[0];
  if (lm == 0) {  // lanes 0 (rows 0..7) and 16 (rows 8..15) of each wave
#pragma unroll
    for (int r = 0; r < 8; ++r) {
      int rl = hi * 8 + r;
      size_t grow = rowBase + wave * 16 + rl;
      float z = gl[r] + bias2;
      float gate = 1.f / (1.f + expf(-z));
      gate_out[grow] = gate;
      logit_out[grow] = gate * rd[r] * 0.03125f;  // 1/sqrt(1024)
    }
  }
}

// ---------------------------------------------------------------------------
// Per-batch masked, length-decayed softmax -> alpha
// ---------------------------------------------------------------------------
__global__ __launch_bounds__(256) void softmax_kernel(
    const unsigned char* __restrict__ mask,  // [B,S] bool
    const float* __restrict__ logit,         // [B,S]
    const float* __restrict__ lam,           // [1]
    float* __restrict__ alpha)               // [B,S]
{
  __shared__ float red[256];
  __shared__ float s_vl, s_max, s_sum;
  const int b = blockIdx.x, tid = threadIdx.x;
  const unsigned char* mrow = mask + (size_t)b * SSZ;
  const float* lrow = logit + (size_t)b * SSZ;
  float* arow = alpha + (size_t)b * SSZ;

  float c = 0.f;
  for (int s = tid; s < SSZ; s += 256) c += (mrow[s] ? 0.f : 1.f);
  red[tid] = c; __syncthreads();
  for (int st = 128; st > 0; st >>= 1) { if (tid < st) red[tid] += red[tid + st]; __syncthreads(); }
  if (tid == 0) s_vl = red[0];
  __syncthreads();
  const float vl = s_vl;
  const float pl = log1pf(expf(lam[0]));  // softplus

  float mx = -3.4e38f;
  for (int s = tid; s < SSZ; s += 256) {
    float dv = mrow[s] ? -10000.f
                       : lrow[s] - pl * fmaxf(vl - 1.f - (float)s, 0.f);
    arow[s] = dv;
    mx = fmaxf(mx, dv);
  }
  red[tid] = mx; __syncthreads();
  for (int st = 128; st > 0; st >>= 1) { if (tid < st) red[tid] = fmaxf(red[tid], red[tid + st]); __syncthreads(); }
  if (tid == 0) s_max = red[0];
  __syncthreads();
  mx = s_max;

  float sum = 0.f;
  for (int s = tid; s < SSZ; s += 256) {
    float e = expf(arow[s] - mx);
    arow[s] = e;
    sum += e;
  }
  red[tid] = sum; __syncthreads();
  for (int st = 128; st > 0; st >>= 1) { if (tid < st) red[tid] += red[tid + st]; __syncthreads(); }
  if (tid == 0) s_sum = red[0];
  __syncthreads();
  const float inv = 1.f / s_sum;
  for (int s = tid; s < SSZ; s += 256) arow[s] *= inv;
}

// ---------------------------------------------------------------------------
// scene[b,d] = sum_s alpha * gate * x   (second streaming pass over x)
// ---------------------------------------------------------------------------
__global__ __launch_bounds__(256) void pool_kernel(
    const float* __restrict__ x,
    const float* __restrict__ alpha,  // [B*S]
    const float* __restrict__ gate,   // [B*S]
    float* __restrict__ out)          // [B,D]
{
  const int b = blockIdx.x;
  const int s0 = blockIdx.y * 128;
  const int tid = threadIdx.x;
  float a0 = 0.f, a1 = 0.f, a2 = 0.f, a3 = 0.f;
  for (int s = s0; s < s0 + 128; ++s) {
    const size_t row = (size_t)b * SSZ + s;
    const float w = alpha[row] * gate[row];
    const float* xr = x + row * DSZ;
    a0 += w * xr[tid];
    a1 += w * xr[tid + 256];
    a2 += w * xr[tid + 512];
    a3 += w * xr[tid + 768];
  }
  atomicAdd(&out[b * DSZ + tid], a0);
  atomicAdd(&out[b * DSZ + tid + 256], a1);
  atomicAdd(&out[b * DSZ + tid + 512], a2);
  atomicAdd(&out[b * DSZ + tid + 768], a3);
}

// ---------------------------------------------------------------------------
extern "C" void kernel_launch(void* const* d_in, const int* in_sizes, int n_in,
                              void* d_out, int out_size, void* d_ws, size_t ws_size,
                              hipStream_t stream) {
  (void)in_sizes; (void)n_in; (void)out_size; (void)ws_size;
  const float* x           = (const float*)d_in[0];
  const unsigned char* msk = (const unsigned char*)d_in[1];
  const float* w1          = (const float*)d_in[2];
  const float* b1          = (const float*)d_in[3];
  const float* w2          = (const float*)d_in[4];
  const float* b2          = (const float*)d_in[5];
  const float* lam         = (const float*)d_in[6];
  const float* query       = (const float*)d_in[7];
  const float* key_w       = (const float*)d_in[8];
  // d_in[9] (key_b) shifts every logit equally -> cancels in softmax; unused.

  char* ws = (char*)d_ws;
  _Float16* w1t   = (_Float16*)ws;                                   // 1 MB
  float* qprime   = (float*)(ws + (1 << 20));                        // 4 KB
  float* gate_ws  = (float*)(ws + (1 << 20) + (4 << 10));            // 256 KB
  float* logit_ws = (float*)(ws + (1 << 20) + (4 << 10) + (256 << 10));
  float* alpha_ws = (float*)(ws + (1 << 20) + (4 << 10) + (512 << 10));

  qprime_kernel<<<DSZ / 128, 128, 0, stream>>>(key_w, query, qprime);
  w1t_kernel<<<(NHSZ * DSZ) / 256, 256, 0, stream>>>(w1, w1t);
  gate_kernel<<<ROWS / MBR, 128, 0, stream>>>(x, w1t, b1, w2, b2, qprime,
                                              gate_ws, logit_ws);
  softmax_kernel<<<BSZ, 256, 0, stream>>>(msk, logit_ws, lam, alpha_ws);
  hipMemsetAsync(d_out, 0, (size_t)BSZ * DSZ * sizeof(float), stream);
  pool_kernel<<<dim3(BSZ, SSZ / 128), 256, 0, stream>>>(x, alpha_ws, gate_ws,
                                                        (float*)d_out);
}
